// GptOssTopKRouter_30236569763902
// MI455X (gfx1250) — matrix operand order repaired and checked by reading, not compiled, not run on hardware
//
#include <hip/hip_runtime.h>
#include <hip/hip_bf16.h>
#include <stdint.h>

// ---------------------------------------------------------------------------
// GptOssTopKRouter for MI455X (gfx1250, wave32, WMMA)
//
// logits = hidden_states[16384,2880] @ router_kernel[2880,128] + bias
// scores = scatter(softmax(top4(logits)))
//
// bf16x3 split GEMM (hi*hi + hi*lo + lo*hi) on v_wmma_f32_16x16x32_bf16.
// A-chunks are converted fp32->bf16(hi/lo) ONCE per block into LDS
// (double-buffered, pre-swizzled A-fragment layout), then all 8 waves
// ds_load their operands — removes the 8x redundant VALU conversion that
// dominated the previous version.  Truncation splits + v_perm_b32 packing
// keep conversion at ~3 VALU/element.
// ---------------------------------------------------------------------------

#define TOKENS   16384
#define HIDDEN   2880
#define NE       128
#define KCH      32
#define NCHUNK   (HIDDEN / KCH)   // 90
#define NTILE    (NE / 16)        // 8
#define FRAGS    (NCHUNK * NTILE) // 720
#define FRAG_ELE 512              // 32 lanes * 16 bf16 per fragment

typedef __bf16 bf16_t;
typedef bf16_t v16bf __attribute__((ext_vector_type(16)));
typedef float  v8f   __attribute__((ext_vector_type(8)));

static __device__ __forceinline__ unsigned fasu(float f) {
  union { float f; unsigned u; } x; x.f = f; return x.u;
}
static __device__ __forceinline__ float uasf(unsigned u) {
  union { unsigned u; float f; } x; x.u = u; return x.f;
}
static __device__ __forceinline__ unsigned short f32_to_bf16_rne(float f) {
  unsigned u = fasu(f);
  unsigned r = u + 0x7fffu + ((u >> 16) & 1u);
  return (unsigned short)(r >> 16);
}

// ---------------------------------------------------------------------------
// Kernel 1 (runs once, 1.47 MB): repack router_kernel [HIDDEN][NE] f32 into
// WMMA B-operand fragments (bf16 hi + lo).  16-bit B (K=32) k order used here (contiguous, same as A):
//   N = lane & 15 ; element e holds K = e + (lane>=16 ? 16 : 0)
// ---------------------------------------------------------------------------
__global__ __launch_bounds__(256) void pack_router_b(
    const float* __restrict__ rk,
    unsigned short* __restrict__ bhi,
    unsigned short* __restrict__ blo) {
  const int wave = threadIdx.x >> 5, lane = threadIdx.x & 31;
  const int frag = blockIdx.x * 8 + wave;   // [0, 720)
  const int chunk = frag >> 3, tile = frag & 7;
  const int n    = lane & 15;
  const int koff = (lane >> 4) << 4;        // 0 or 16
  const float* src = rk + (size_t)(chunk * KCH + koff) * NE + tile * 16 + n;

  unsigned short hi[16], lo[16];
#pragma unroll
  for (int e = 0; e < 16; ++e) {
    float f = src[(size_t)e * NE];
    unsigned short h = f32_to_bf16_rne(f);
    hi[e] = h;
    lo[e] = f32_to_bf16_rne(f - uasf(((unsigned)h) << 16));
  }
  const size_t base = (size_t)(frag * 32 + lane) * 16;  // ushort idx, 32B aligned
  uint4* dh = (uint4*)(bhi + base);
  uint4* dl = (uint4*)(blo + base);
  dh[0] = *(const uint4*)&hi[0]; dh[1] = *(const uint4*)&hi[8];
  dl[0] = *(const uint4*)&lo[0]; dl[1] = *(const uint4*)&lo[8];
}

// ---------------------------------------------------------------------------
// Kernel 2: GEMM.  Block = 256 threads = 8 waves = 32 tokens x 128 experts.
// Wave w owns expert tile w and both 16-token M tiles.
//
// LDS A staging (per chunk slot, 4 KB; 2 slots): for m-tile m, lane L:
//   bytes [m*2048 + L*64 +  0 .. 31] = hi bf16 fragment (16 elems)
//   bytes [m*2048 + L*64 + 32 .. 63] = lo bf16 fragment
// A-fragment element map (16-bit A 16x32): M = L&15, base=(L>=16?16:0),
//   e: K=k0+base+e (contiguous)
// Producer thread t: row = t>>3, covers K = k0 + (t&7)*4 .. +3  (one float4),
// which lands as 4 consecutive elements of one fragment half -> 1 b64 store.
// ---------------------------------------------------------------------------
union Frag32 { uint4 q[2]; v16bf v; };

__global__ __launch_bounds__(256) void router_gemm(
    const float* __restrict__ A,
    const unsigned short* __restrict__ bhi,
    const unsigned short* __restrict__ blo,
    const float* __restrict__ bias,
    float* __restrict__ logits) {
  __shared__ __attribute__((aligned(16))) unsigned char smem[2 * 4096];

  const int lane = threadIdx.x & 31;
  const int tile = threadIdx.x >> 5;        // expert tile 0..7
  const int tokBase = blockIdx.x * 32;

  // ---- producer-role constants (each thread converts 4 elements/chunk) ----
  const int prow = threadIdx.x >> 3;          // token row 0..31
  const int pkq  = (threadIdx.x & 7) * 4;     // K offset within chunk
  const int pe0  = pkq & 15;
  const int planeF = (((pkq >> 4) & 1) << 4) + (prow & 15);
  const unsigned psoff = (unsigned)((prow >> 4) * 2048 + planeF * 64 + pe0 * 2);
  const float* pA = A + (size_t)(tokBase + prow) * HIDDEN + pkq;

  // ---- consumer-role constants ----
  const int abase = (lane >> 4) << 3;         // 0 or 8 (C-layout M offset)
  const unsigned short* ph = bhi + ((size_t)tile * 32 + lane) * 16;
  const unsigned short* pl = blo + ((size_t)tile * 32 + lane) * 16;
  const size_t fragStride = (size_t)NTILE * FRAG_ELE;   // ushorts per chunk

  v8f acc0 = {};
  v8f acc1 = {};

  // ---- producer: fp32 -> bf16 hi/lo (truncation split) into LDS slot ----
  auto produce = [&](int c) {
    const float4 a = *(const float4*)(pA + (size_t)c * KCH);
    const unsigned u0 = fasu(a.x), u1 = fasu(a.y), u2 = fasu(a.z), u3 = fasu(a.w);
    // pack high halves of two dwords in one v_perm_b32
    const unsigned hp0 = __builtin_amdgcn_perm(u1, u0, 0x07060302u);
    const unsigned hp1 = __builtin_amdgcn_perm(u3, u2, 0x07060302u);
    const float l0 = a.x - uasf(u0 & 0xffff0000u);
    const float l1 = a.y - uasf(u1 & 0xffff0000u);
    const float l2 = a.z - uasf(u2 & 0xffff0000u);
    const float l3 = a.w - uasf(u3 & 0xffff0000u);
    const unsigned lp0 = __builtin_amdgcn_perm(fasu(l1), fasu(l0), 0x07060302u);
    const unsigned lp1 = __builtin_amdgcn_perm(fasu(l3), fasu(l2), 0x07060302u);
    unsigned char* base = smem + (unsigned)(c & 1) * 4096 + psoff;
    uint2 h; h.x = hp0; h.y = hp1;
    uint2 l; l.x = lp0; l.y = lp1;
    *(uint2*)(base)      = h;   // ds_store_b64
    *(uint2*)(base + 32) = l;   // ds_store_b64
  };

  produce(0);
  __syncthreads();

  for (int c = 0; c < NCHUNK; ++c) {
    if (c + 1 < NCHUNK) produce(c + 1);

    // ---- consume A fragments from LDS (4x ds_load_b128 per m-tile) ----
    const unsigned char* s = smem + (unsigned)(c & 1) * 4096;
    Frag32 ah0, al0, ah1, al1;
    {
      const uint4* p0 = (const uint4*)(s + lane * 64);
      ah0.q[0] = p0[0]; ah0.q[1] = p0[1];
      al0.q[0] = p0[2]; al0.q[1] = p0[3];
      const uint4* p1 = (const uint4*)(s + 2048 + lane * 64);
      ah1.q[0] = p1[0]; ah1.q[1] = p1[1];
      al1.q[0] = p1[2]; al1.q[1] = p1[3];
    }

    // ---- B fragments (pre-swizzled bf16 in L2, contiguous 32B/lane) ----
    Frag32 uh, ul;
    {
      const uint4* hq = (const uint4*)ph;
      const uint4* lq = (const uint4*)pl;
      uh.q[0] = hq[0]; uh.q[1] = hq[1];
      ul.q[0] = lq[0]; ul.q[1] = lq[1];
    }
    ph += fragStride; pl += fragStride;

    // ---- bf16x3: hi*hi + hi*lo + lo*hi, two independent accumulators ----
    acc0 = __builtin_amdgcn_wmma_f32_16x16x32_bf16(false, ah0.v, false, uh.v,
                                                   (short)0, acc0, false, false);
    acc1 = __builtin_amdgcn_wmma_f32_16x16x32_bf16(false, ah1.v, false, uh.v,
                                                   (short)0, acc1, false, false);
    acc0 = __builtin_amdgcn_wmma_f32_16x16x32_bf16(false, ah0.v, false, ul.v,
                                                   (short)0, acc0, false, false);
    acc1 = __builtin_amdgcn_wmma_f32_16x16x32_bf16(false, ah1.v, false, ul.v,
                                                   (short)0, acc1, false, false);
    acc0 = __builtin_amdgcn_wmma_f32_16x16x32_bf16(false, al0.v, false, uh.v,
                                                   (short)0, acc0, false, false);
    acc1 = __builtin_amdgcn_wmma_f32_16x16x32_bf16(false, al1.v, false, uh.v,
                                                   (short)0, acc1, false, false);

    __syncthreads();   // slot c&1 free for produce(c+2); produce(c+1) visible
  }

  // Epilogue: C/D layout — VGPR r: M = r + (lane>=16?8:0), N = lane&15
  const int expert = tile * 16 + (lane & 15);
  const float b = bias[expert];
#pragma unroll
  for (int r = 0; r < 8; ++r) {
    const int t0 = tokBase + r + abase;
    logits[(size_t)t0 * NE + expert]        = acc0[r] + b;
    logits[(size_t)(t0 + 16) * NE + expert] = acc1[r] + b;
  }
}

// ---------------------------------------------------------------------------
// Kernel 3: one wave per token.  Coalesced float4 load of 128 logits, 4
// rounds of wave argmax (sortable-uint key, lowest-index tie-break like
// lax.top_k), softmax over the winners, coalesced float4 row write.
// ---------------------------------------------------------------------------
__global__ __launch_bounds__(256) void topk_softmax(
    const float* __restrict__ logits, float* __restrict__ out) {
  const int wave = threadIdx.x >> 5, lane = threadIdx.x & 31;
  const int token = blockIdx.x * 8 + wave;

  const float4 v = ((const float4*)(logits + (size_t)token * NE))[lane];
  const float vals[4] = {v.x, v.y, v.z, v.w};
  unsigned keys[4];
#pragma unroll
  for (int j = 0; j < 4; ++j) {
    unsigned u = fasu(vals[j]);
    keys[j] = (u & 0x80000000u) ? ~u : (u | 0x80000000u);
  }

  bool used[4] = {false, false, false, false};
  int selIdx[4]; float selVal[4];
#pragma unroll
  for (int k = 0; k < 4; ++k) {
    unsigned long long best = 0ull;
#pragma unroll
    for (int j = 0; j < 4; ++j) {
      if (!used[j]) {
        unsigned long long cand =
            ((unsigned long long)keys[j] << 32) | (unsigned)(127 - (lane * 4 + j));
        best = (cand > best) ? cand : best;
      }
    }
#pragma unroll
    for (int s = 16; s > 0; s >>= 1) {  // wave32 max-reduce
      unsigned hi = (unsigned)(best >> 32), lo = (unsigned)best;
      unsigned ohi = __shfl_xor(hi, s, 32);
      unsigned olo = __shfl_xor(lo, s, 32);
      unsigned long long o = ((unsigned long long)ohi << 32) | olo;
      best = (o > best) ? o : best;
    }
    const int idx = 127 - (int)(best & 0xffu);
    const unsigned key = (unsigned)(best >> 32);
    selIdx[k] = idx;
    selVal[k] = uasf((key & 0x80000000u) ? (key ^ 0x80000000u) : ~key);
    if ((idx >> 2) == lane) used[idx & 3] = true;
  }

  const float m = selVal[0];  // round 0 winner is the global max
  const float e0 = __expf(selVal[0] - m), e1 = __expf(selVal[1] - m),
              e2 = __expf(selVal[2] - m), e3 = __expf(selVal[3] - m);
  const float inv = 1.0f / (e0 + e1 + e2 + e3);
  const float w[4] = {e0 * inv, e1 * inv, e2 * inv, e3 * inv};

  float o[4] = {0.f, 0.f, 0.f, 0.f};
#pragma unroll
  for (int k = 0; k < 4; ++k)
    if ((selIdx[k] >> 2) == lane) o[selIdx[k] & 3] = w[k];
  float4 ov; ov.x = o[0]; ov.y = o[1]; ov.z = o[2]; ov.w = o[3];
  ((float4*)(out + (size_t)token * NE))[lane] = ov;
}

// ---------------------------------------------------------------------------
// Launch.  Workspace (needs ~9.9 MB):
//   [0, 720KB)        packed B hi  (720 frags * 512 bf16)
//   [720KB, 1440KB)   packed B lo
//   [1440KB, +8.4MB)  fp32 logits [16384,128]
// ---------------------------------------------------------------------------
extern "C" void kernel_launch(void* const* d_in, const int* in_sizes, int n_in,
                              void* d_out, int out_size, void* d_ws, size_t ws_size,
                              hipStream_t stream) {
  (void)in_sizes; (void)n_in; (void)out_size; (void)ws_size;
  const float* hs   = (const float*)d_in[0];
  const float* rk   = (const float*)d_in[1];
  const float* bias = (const float*)d_in[2];
  float* out        = (float*)d_out;

  unsigned short* bhi = (unsigned short*)d_ws;
  unsigned short* blo = bhi + (size_t)FRAGS * FRAG_ELE;
  float* logits = (float*)(blo + (size_t)FRAGS * FRAG_ELE);

  pack_router_b<<<NCHUNK, 256, 0, stream>>>(rk, bhi, blo);                  // 90 blocks
  router_gemm<<<TOKENS / 32, 256, 0, stream>>>(hs, bhi, blo, bias, logits); // 512 blocks
  topk_softmax<<<TOKENS / 8, 256, 0, stream>>>(logits, out);                // 2048 blocks
}
